// YemongDynamics_34437047779442
// MI455X (gfx1250) — matrix operand order repaired and checked by reading, not compiled, not run on hardware
//
#include <hip/hip_runtime.h>
#include <math.h>

// ---- problem constants ------------------------------------------------------
#define NB_ 8          // batch
#define NS_ 512        // seq len
#define NN_ 8          // ships
#define NT_ (NB_*NS_*NN_)   // 32768 tokens (token order: t = (b*NN_+n)*NS_+s)
#define DM_ 256
#define DI_ 512
#define DST_ 8
#define NLAY_ 4
#define KCAT_ 320      // 304 concat width zero-padded to a multiple of 32

// output offsets (floats) in d_out: next_state, logits, value, reward, Z
#define OFF_NS_  0L
#define OFF_AL_  163840L
#define OFF_VAL_ 557056L
#define OFF_REW_ 561152L
#define OFF_Z_   565248L

typedef _Float16 f16;
typedef __attribute__((ext_vector_type(16))) _Float16 v16h;
typedef __attribute__((ext_vector_type(8)))  _Float16 v8h;
typedef __attribute__((ext_vector_type(8)))  float    v8f;

__device__ __forceinline__ float siluf(float x){ return x / (1.f + __expf(-x)); }

// ---- operand packing --------------------------------------------------------
// cast a f32 activation image to f16 (same row-major layout)
__global__ void k_cast16(const float* __restrict__ in, f16* __restrict__ out, long n)
{
  long i = (long)blockIdx.x * blockDim.x + threadIdx.x;
  if (i < n) out[i] = (f16)in[i];
}
// pack weight into [N, Kp] f16 (k-contiguous), folding optional transpose + K pad
__global__ void k_pack_w(const float* __restrict__ W, f16* __restrict__ pw,
                         int N, int Kp, int Ko, int TW)
{
  long i = (long)blockIdx.x * blockDim.x + threadIdx.x;
  if (i >= (long)N * Kp) return;
  int k = i % Kp; int n = i / Kp;
  float v = 0.f;
  if (k < Ko) v = TW ? W[(long)n * Ko + k] : W[(long)k * N + n];
  pw[i] = (f16)v;
}

// ---- WMMA GEMM:  C[M,N] = act(A@W + bias) (+res) ---------------------------
// A: f16 [M,K] row-major.  Wp: f16 [N,K] k-contiguous.  M%64==0, N%16==0, K%32==0.
// One wave: 16x64 tile = 4x v_wmma_f32_16x16x32_f16; block = 64x128.
__global__ __launch_bounds__(256)
void k_gemm(const f16* __restrict__ A, const f16* __restrict__ Wp,
            const float* __restrict__ bias, const float* __restrict__ res,
            float* __restrict__ C, int M, int N, int K, int ACT)
{
  const int lane  = threadIdx.x & 31;
  const int wave  = threadIdx.x >> 5;
  const int tileM = blockIdx.x * 64 + (wave >> 1) * 16;
  const int tileN = blockIdx.y * 128 + (wave & 1) * 64;
  if (tileN >= N) return;                       // uniform per wave
  const int l16   = lane & 15;
  const int nhalf = lane >> 4;

  // A fragment rows / B fragment columns for this lane
  const f16* arow = A + (long)(tileM + l16) * K + nhalf * 16;
  const f16* bp0  = Wp + (long)(tileN +  0 + l16) * K + nhalf * 16;
  const f16* bp1  = Wp + (long)(tileN + 16 + l16) * K + nhalf * 16;
  const f16* bp2  = Wp + (long)(tileN + 32 + l16) * K + nhalf * 16;
  const f16* bp3  = Wp + (long)(tileN + 48 + l16) * K + nhalf * 16;
  const bool ok1 = (tileN + 16) < N, ok2 = (tileN + 32) < N, ok3 = (tileN + 48) < N;

  v8f acc[4];
  #pragma unroll
  for (int t = 0; t < 4; ++t)
    #pragma unroll
    for (int r = 0; r < 8; ++r) acc[t][r] = 0.f;

  for (int kk = 0; kk < K; kk += 32) {
    // A fragment: one contiguous 32B run per lane (k = 16*half + s, matches B)
    v8h a0 = *(const v8h*)(arow + kk);
    v8h a1 = *(const v8h*)(arow + kk + 8);
    v16h af = __builtin_shufflevector(a0, a1, 0,1,2,3,4,5,6,7,8,9,10,11,12,13,14,15);
    // B fragments: one contiguous 32B run per lane (k = 16*half + s, same as A)
    {
      v16h bf = *(const v16h*)(bp0 + kk);
      acc[0] = __builtin_amdgcn_wmma_f32_16x16x32_f16(false, af, false, bf, (short)0, acc[0], false, false);
    }
    if (ok1) {
      v16h bf = *(const v16h*)(bp1 + kk);
      acc[1] = __builtin_amdgcn_wmma_f32_16x16x32_f16(false, af, false, bf, (short)0, acc[1], false, false);
    }
    if (ok2) {
      v16h bf = *(const v16h*)(bp2 + kk);
      acc[2] = __builtin_amdgcn_wmma_f32_16x16x32_f16(false, af, false, bf, (short)0, acc[2], false, false);
    }
    if (ok3) {
      v16h bf = *(const v16h*)(bp3 + kk);
      acc[3] = __builtin_amdgcn_wmma_f32_16x16x32_f16(false, af, false, bf, (short)0, acc[3], false, false);
    }
  }
  // epilogue (C/D layout: lanes 0-15: M=r, N=lane; lanes 16-31: M=8+r, N=lane-16)
  #pragma unroll
  for (int t = 0; t < 4; ++t) {
    int ncol = tileN + t * 16 + l16;
    if (ncol >= N) continue;                    // uniform per 16-subtile
    float bv = bias ? bias[ncol] : 0.f;
    #pragma unroll
    for (int r = 0; r < 8; ++r) {
      int m = tileM + (nhalf << 3) + r;
      float v = acc[t][r] + bv;
      if (ACT == 1) v = siluf(v);
      if (res) v += res[(long)m * N + ncol];
      C[(long)m * N + ncol] = v;
    }
  }
}

// ---- state encoder layer 1 (K=12) + alive mask ------------------------------
__global__ void k_state_enc(const float* __restrict__ state, const float* __restrict__ dead,
                            const float* __restrict__ W1, const float* __restrict__ b1,
                            int* __restrict__ albuf, float* __restrict__ out)
{
  long idx = (long)blockIdx.x * blockDim.x + threadIdx.x;
  if (idx >= (long)NT_ * DM_) return;
  int o = idx & 255; long t = idx >> 8;
  int s = t % NS_; long bn = t / NS_; int n = bn % NN_; int b = bn / NN_;
  long base = (((long)b * NS_ + s) * NN_ + n) * 12;
  bool alive = state[base] > 0.f;
  if (o == 0) albuf[(b * NS_ + s) * NN_ + n] = alive ? 1 : 0;
  float acc = b1[o];
  #pragma unroll
  for (int c = 0; c < 12; ++c) {
    float sv = alive ? state[base + c] : dead[c];
    acc += sv * W1[c * DM_ + o];
  }
  out[t * DM_ + o] = siluf(acc);
}

// ---- build concat [s + team + ship + rst*reset , act_embed(prev), pad] [T,320]
__global__ void k_build_cat(const float* __restrict__ slin, const int* __restrict__ team_ids,
                            const int* __restrict__ seq_idx, const int* __restrict__ prev_action,
                            const float* __restrict__ team_emb, const float* __restrict__ ship_emb,
                            const float* __restrict__ resetv,
                            const float* __restrict__ ai_p, const float* __restrict__ ai_t,
                            const float* __restrict__ ai_s, float* __restrict__ cat)
{
  long idx = (long)blockIdx.x * blockDim.x + threadIdx.x;
  if (idx >= (long)NT_ * KCAT_) return;
  int c = idx % KCAT_; long t = idx / KCAT_;
  int s = t % NS_; long bn = t / NS_; int n = bn % NN_; int b = bn / NN_;
  float v = 0.f;
  if (c < 256) {
    int rst = (s > 0 && seq_idx[b * NS_ + s] != seq_idx[b * NS_ + s - 1]) ? 1 : 0;
    v = slin[t * DM_ + c]
      + team_emb[team_ids[b * NN_ + n] * DM_ + c]
      + ship_emb[n * DM_ + c]
      + (rst ? resetv[c] : 0.f);
  } else if (c < 304) {
    long ab = (((long)b * NS_ + s) * NN_ + n) * 3;
    int c2 = c - 256;
    if (c2 < 16)       v = ai_p[prev_action[ab + 0] * 16 + c2];
    else if (c2 < 32)  v = ai_t[prev_action[ab + 1] * 16 + (c2 - 16)];
    else               v = ai_s[prev_action[ab + 2] * 16 + (c2 - 32)];
  }
  cat[t * KCAT_ + c] = v;
}

// ---- row norm over 256: mode 0 = RMS(+gain), mode 1 = LayerNorm -------------
__global__ __launch_bounds__(256)
void k_norm(const float* __restrict__ x, const float* __restrict__ g,
            const float* __restrict__ beta, const float* __restrict__ res,
            float* __restrict__ out, long rows, int mode, int do_silu)
{
  int lane = threadIdx.x & 31;
  long row = (long)blockIdx.x * 8 + (threadIdx.x >> 5);
  if (row >= rows) return;
  const float* xr = x + row * DM_;
  float v[8], ss = 0.f, sm = 0.f;
  #pragma unroll
  for (int i = 0; i < 8; ++i) { v[i] = xr[lane + i * 32]; ss += v[i] * v[i]; sm += v[i]; }
  for (int off = 16; off; off >>= 1) { ss += __shfl_xor(ss, off, 32); sm += __shfl_xor(sm, off, 32); }
  float scale, shift = 0.f;
  if (mode == 0) scale = rsqrtf(ss / 256.f + 1e-6f);
  else { float mu = sm / 256.f; scale = rsqrtf(ss / 256.f - mu * mu + 1e-5f); shift = mu; }
  #pragma unroll
  for (int i = 0; i < 8; ++i) {
    int c = lane + i * 32;
    float y = (v[i] - shift) * scale * g[c] + (beta ? beta[c] : 0.f);
    if (do_silu) y = siluf(y);
    if (res) y += res[row * DM_ + c];
    out[row * DM_ + c] = y;
  }
}

// ---- depthwise causal conv (episode-masked) + SiLU --------------------------
__global__ void k_conv(const float* __restrict__ xz, const float* __restrict__ cw,
                       const float* __restrict__ cb, const int* __restrict__ seq_idx,
                       float* __restrict__ u)
{
  long idx = (long)blockIdx.x * blockDim.x + threadIdx.x;
  if (idx >= (long)NT_ * DI_) return;
  int d = idx % DI_; long t = idx / DI_;
  int s = t % NS_; long bn = t / NS_; int b = bn / NN_;
  int mysi = seq_idx[b * NS_ + s];
  float acc = cb[d];
  #pragma unroll
  for (int k = 0; k < 4; ++k) {
    int sp = s - (3 - k);
    if (sp >= 0 && seq_idx[b * NS_ + sp] == mysi)
      acc += xz[(bn * NS_ + sp) * 1024 + d] * cw[d * 4 + k];
  }
  u[t * DI_ + d] = siluf(acc);
}

// ---- dt = softplus(dtr @ W_dt + b_dt) ---------------------------------------
__global__ void k_dt(const float* __restrict__ dbl, const float* __restrict__ Wdt,
                     const float* __restrict__ bdt, float* __restrict__ dt)
{
  long idx = (long)blockIdx.x * blockDim.x + threadIdx.x;
  if (idx >= (long)NT_ * DI_) return;
  int d = idx % DI_; long t = idx / DI_;
  float acc = bdt[d];
  #pragma unroll
  for (int r = 0; r < 16; ++r) acc += dbl[t * 32 + r] * Wdt[r * DI_ + d];
  dt[t * DI_ + d] = (acc > 20.f) ? acc : log1pf(__expf(acc));
}

// ---- selective scan: thread = (sequence, channel), state in registers -------
__global__ void k_scan(const float* __restrict__ dt, const float* __restrict__ u,
                       const float* __restrict__ dbl, const float* __restrict__ xz,
                       const float* __restrict__ Alog, const float* __restrict__ Dp,
                       const int* __restrict__ seq_idx, float* __restrict__ y)
{
  int tid = blockIdx.x * blockDim.x + threadIdx.x;
  if (tid >= NB_ * NN_ * DI_) return;
  int d = tid & (DI_ - 1); int bn = tid >> 9; int b = bn >> 3;
  float Arow[DST_], h[DST_];
  #pragma unroll
  for (int st = 0; st < DST_; ++st) { Arow[st] = -__expf(Alog[d * DST_ + st]); h[st] = 0.f; }
  float Dd = Dp[d];
  const int* si = seq_idx + b * NS_;
  for (int s = 0; s < NS_; ++s) {
    long t = (long)bn * NS_ + s;
    if (s == 0 || si[s] != si[s - 1]) {
      #pragma unroll
      for (int st = 0; st < DST_; ++st) h[st] = 0.f;
    }
    float dtv = dt[t * DI_ + d], uv = u[t * DI_ + d];
    float du = dtv * uv, yv = 0.f;
    #pragma unroll
    for (int st = 0; st < DST_; ++st) {
      h[st] = __expf(dtv * Arow[st]) * h[st] + du * dbl[t * 32 + 16 + st];
      yv += h[st] * dbl[t * 32 + 24 + st];
    }
    yv += uv * Dd;
    float z = xz[t * 1024 + DI_ + d];
    y[t * DI_ + d] = yv * siluf(z);
  }
}

// ---- relational trunk -> per-layer attention bias [4][B,S,N,N,8] ------------
__global__ __launch_bounds__(64)
void k_trunk(const float* __restrict__ pos, const float* __restrict__ vel,
             const float* __restrict__ Wr1, const float* __restrict__ br1,
             const float* __restrict__ Wr2, const float* __restrict__ br2,
             const float* __restrict__ adW, const float* __restrict__ adb,
             float* __restrict__ biasb)
{
  __shared__ float f[5], h1[64], h2[64];
  long pb = blockIdx.x;                     // (b,s,i,j)
  int j = pb & 7, i = (pb >> 3) & 7, s = (pb >> 6) & (NS_ - 1), b = (int)(pb >> 15);
  int tid = threadIdx.x;
  if (tid == 0) {
    long pi = (((long)b * NS_ + s) * NN_ + i) * 2;
    long pj = (((long)b * NS_ + s) * NN_ + j) * 2;
    float dx = pos[pi] - pos[pj];     dx -= 1024.f * roundf(dx * (1.f / 1024.f));
    float dy = pos[pi + 1] - pos[pj + 1]; dy -= 1024.f * roundf(dy * (1.f / 1024.f));
    float dvx = vel[pi] - vel[pj], dvy = vel[pi + 1] - vel[pj + 1];
    float dist = sqrtf(dx * dx + dy * dy + 1e-8f);
    f[0] = dx * (1.f / 1024.f); f[1] = dy * (1.f / 1024.f);
    f[2] = dvx; f[3] = dvy; f[4] = dist * (1.f / 1024.f);
  }
  __syncthreads();
  float a = br1[tid];
  #pragma unroll
  for (int c = 0; c < 5; ++c) a += f[c] * Wr1[c * 64 + tid];
  h1[tid] = siluf(a);
  __syncthreads();
  float a2 = br2[tid];
  for (int c = 0; c < 64; ++c) a2 += h1[c] * Wr2[c * 64 + tid];
  h2[tid] = siluf(a2);
  __syncthreads();
  if (tid < 32) {
    int l = tid >> 3, h = tid & 7;
    float bb = adb[l * 8 + h];
    for (int c = 0; c < 64; ++c) bb += h2[c] * adW[(l * 64 + c) * 8 + h];
    biasb[((long)l * NB_ * NS_ * 64 + pb) * 8 + h] = bb;
  }
}

// ---- ship attention (8 keys, 8 heads of 32) with relational bias ------------
__global__ void k_attn(const float* __restrict__ q, const float* __restrict__ k,
                       const float* __restrict__ v, const float* __restrict__ bias,
                       const int* __restrict__ albuf, float* __restrict__ o)
{
  int tid = blockIdx.x * blockDim.x + threadIdx.x;
  if (tid >= NB_ * NS_ * 8 * 8) return;
  int i = tid & 7, h = (tid >> 3) & 7, s = (tid >> 6) & (NS_ - 1), b = tid >> 15;
  long ti = ((long)(b * NN_ + i)) * NS_ + s;
  float qr[32];
  #pragma unroll
  for (int c = 0; c < 32; ++c) qr[c] = q[ti * DM_ + h * 32 + c];
  float sc[8], mx = -1e30f;
  for (int j = 0; j < 8; ++j) {
    long tj = ((long)(b * NN_ + j)) * NS_ + s;
    float a = 0.f;
    #pragma unroll
    for (int c = 0; c < 32; ++c) a += qr[c] * k[tj * DM_ + h * 32 + c];
    a *= 0.1767766953f;   // 1/sqrt(32)
    a += bias[((((long)(b * NS_ + s)) * NN_ + i) * NN_ + j) * 8 + h];
    if (!albuf[(b * NS_ + s) * NN_ + j]) a = -1e9f;
    sc[j] = a; mx = fmaxf(mx, a);
  }
  float sum = 0.f;
  #pragma unroll
  for (int j = 0; j < 8; ++j) { sc[j] = __expf(sc[j] - mx); sum += sc[j]; }
  float inv = 1.f / sum;
  float orr[32];
  #pragma unroll
  for (int c = 0; c < 32; ++c) orr[c] = 0.f;
  for (int j = 0; j < 8; ++j) {
    long tj = ((long)(b * NN_ + j)) * NS_ + s;
    float w = sc[j] * inv;
    #pragma unroll
    for (int c = 0; c < 32; ++c) orr[c] += w * v[tj * DM_ + h * 32 + c];
  }
  #pragma unroll
  for (int c = 0; c < 32; ++c) o[ti * DM_ + h * 32 + c] = orr[c];
}

// ---- generic narrow head: C[M,Ns] = act(A@W + b), optional (b,n,s)->(b,s,n) -
__global__ void k_small(const float* __restrict__ A, const float* __restrict__ W,
                        const float* __restrict__ bias, float* __restrict__ out,
                        long Mrows, int K, int Ns, int do_silu, int reorder)
{
  long idx = (long)blockIdx.x * blockDim.x + threadIdx.x;
  if (idx >= Mrows * Ns) return;
  int o = idx % Ns; long r = idx / Ns;
  float acc = bias ? bias[o] : 0.f;
  const float* ar = A + r * K;
  for (int c = 0; c < K; ++c) acc += ar[c] * W[c * Ns + o];
  if (do_silu) acc = siluf(acc);
  long orow = r;
  if (reorder) {
    int s = r % NS_; long bn = r / NS_; int n = bn % NN_; int b = bn / NN_;
    orow = ((long)(b * NS_ + s)) * NN_ + n;
  }
  out[orow * Ns + o] = acc;
}

// ---- Z copy out in (b,s,n) order --------------------------------------------
__global__ void k_zout(const float* __restrict__ xm, float* __restrict__ out)
{
  long idx = (long)blockIdx.x * blockDim.x + threadIdx.x;
  if (idx >= (long)NT_ * DM_) return;
  int c = idx & 255; long t = idx >> 8;
  int s = t % NS_; long bn = t / NS_; int n = bn % NN_; int b = bn / NN_;
  out[OFF_Z_ + (((long)(b * NS_ + s)) * NN_ + n) * DM_ + c] = xm[t * DM_ + c];
}

// ---- dynamics concat: [Z, act_embed(argmax(logits)), pad] [T,320] -----------
__global__ void k_dyn_cat(const float* __restrict__ Z, const float* __restrict__ logits,
                          const float* __restrict__ adn_p, const float* __restrict__ adn_t,
                          const float* __restrict__ adn_s, float* __restrict__ cat)
{
  long idx = (long)blockIdx.x * blockDim.x + threadIdx.x;
  if (idx >= (long)NT_ * KCAT_) return;
  int c = idx % KCAT_; long t = idx / KCAT_;
  if (c < 256) { cat[t * KCAT_ + c] = Z[t * DM_ + c]; return; }
  if (c >= 304) { cat[t * KCAT_ + c] = 0.f; return; }
  int s = t % NS_; long bn = t / NS_; int n = bn % NN_; int b = bn / NN_;
  const float* lg = logits + (((long)(b * NS_ + s)) * NN_ + n) * 12;
  int c2 = c - 256; float v;
  if (c2 < 16) {
    int a = 0; float bv = lg[0];
    for (int i = 1; i < 3; ++i) if (lg[i] > bv) { bv = lg[i]; a = i; }
    v = adn_p[a * 16 + c2];
  } else if (c2 < 32) {
    int a = 0; float bv = lg[3];
    for (int i = 1; i < 7; ++i) if (lg[3 + i] > bv) { bv = lg[3 + i]; a = i; }
    v = adn_t[a * 16 + (c2 - 16)];
  } else {
    int a = (lg[11] > lg[10]) ? 1 : 0;
    v = adn_s[a * 16 + (c2 - 32)];
  }
  cat[t * KCAT_ + c] = v;
}

// ---- pool query vector (single token) ---------------------------------------
__global__ void k_poolq(const float* __restrict__ tok, const float* __restrict__ in_w,
                        const float* __restrict__ in_b, float* __restrict__ qv)
{
  int o = blockIdx.x * blockDim.x + threadIdx.x;
  if (o >= DM_) return;
  float a = in_b[o];
  for (int c = 0; c < DM_; ++c) a += tok[c] * in_w[o * DM_ + c];
  qv[o] = a;
}

// ---- pooling attention: 4 heads of 64 over 8 keys, dead mask ----------------
__global__ void k_pool_attn(const float* __restrict__ qv, const float* __restrict__ kp,
                            const float* __restrict__ vp, const int* __restrict__ albuf,
                            float* __restrict__ o)
{
  int tid = blockIdx.x * blockDim.x + threadIdx.x;
  if (tid >= NB_ * NS_ * 4) return;
  int h = tid & 3, bt = tid >> 2;
  int b = bt >> 9, s = bt & (NS_ - 1);
  float sc[8], mx = -1e30f;
  for (int n = 0; n < 8; ++n) {
    long tr = ((long)(b * NN_ + n)) * NS_ + s;
    float a = 0.f;
    for (int c = 0; c < 64; ++c) a += qv[h * 64 + c] * kp[tr * DM_ + h * 64 + c];
    a *= 0.125f;  // 1/sqrt(64)
    if (!albuf[(b * NS_ + s) * NN_ + n]) a = -1e9f;
    sc[n] = a; mx = fmaxf(mx, a);
  }
  float sum = 0.f;
  #pragma unroll
  for (int n = 0; n < 8; ++n) { sc[n] = __expf(sc[n] - mx); sum += sc[n]; }
  float inv = 1.f / sum;
  for (int c = 0; c < 64; ++c) {
    float acc = 0.f;
    for (int n = 0; n < 8; ++n) {
      long tr = ((long)(b * NN_ + n)) * NS_ + s;
      acc += sc[n] * inv * vp[tr * DM_ + h * 64 + c];
    }
    o[(long)bt * DM_ + h * 64 + c] = acc;
  }
}

// ============================================================================
extern "C" void kernel_launch(void* const* d_in, const int* in_sizes, int n_in,
                              void* d_out, int out_size, void* d_ws, size_t ws_size,
                              hipStream_t stream) {
  (void)in_sizes; (void)n_in; (void)out_size; (void)ws_size;
  const float* state       = (const float*)d_in[0];
  const float* pos         = (const float*)d_in[1];
  const float* vel         = (const float*)d_in[2];
  const int*   prev_action = (const int*)d_in[3];
  const int*   team_ids    = (const int*)d_in[4];
  const int*   seq_idx     = (const int*)d_in[5];
  const float* const* P    = (const float* const*)(d_in + 6);  // param leaves

  // Fully-sorted (Python key order) param-leaf indices. Uppercase < lowercase,
  // so 'Wr1','Wr2' sort first; 'br1','br2' land between 'blocks' and 'dead'.
  enum {
    P_WR1 = 0, P_WR2,
    P_ACT_W, P_ACT_B, P_AD_W, P_AD_B, P_ADN_P, P_ADN_S, P_ADN_T, P_AI_P, P_AI_S, P_AI_T,
    // blocks.* (stacked over 4 layers)
    B_A_LOG, B_D, B_W_DT, B_W_IN, B_W_OUT, B_W_X, B_WK, B_WO, B_WQ, B_WV,
    B_B_DT, B_BK, B_BO, B_BQ, B_BV, B_CONV_B, B_CONV_W, B_G1, B_G2,
    P_BR1, P_BR2, P_DEAD, P_DYN_W, P_DYN_B, P_DYN_G,
    P_FFN_W1, P_FFN_W2, P_FFN_B1, P_FFN_B2, P_FFN_G,
    P_FUS_W, P_FUS_B, P_FUS_G, P_LNV_B, P_LNV_W,
    PR_IN_B, PR_IN_W, PR_OUT_B, PR_OUT_W, PV_IN_B, PV_IN_W, PV_OUT_B, PV_OUT_W,
    P_RESET, P_RH_W1, P_RH_W2, P_RH_B1, P_RH_B2, P_RN_G,
    P_SE_W1, P_SE_W2, P_SE_B1, P_SE_B2, P_SHIP_EMB, P_TEAM_EMB, P_TOK_REW, P_TOK_VAL,
    P_VH_W1, P_VH_W2, P_VH_B1, P_VH_B2, P_WH_W1, P_WH_W2, P_WH_B1, P_WH_B2
  };

  float* out = (float*)d_out;
  const long T = NT_, BT = (long)NB_ * NS_;

  // workspace layout
  float* w = (float*)d_ws;
  float* xm   = w; w += T * DM_;       // residual stream (token order b,n,s)
  float* tmpA = w; w += T * 1024;      // xz / concat(320)
  float* tmpB = w; w += T * DI_;       // u / ffn hidden / attn out / wh hidden
  float* tmpC = w; w += T * DI_;       // dt / 256-wide intermediates
  float* tmpD = w; w += T * DI_;       // scan y / state-enc linear
  float* tmpN = w; w += T * DM_;       // normed stream
  float* qb   = w; w += T * DM_;
  float* kb   = w; w += T * DM_;
  float* vb   = w; w += T * DM_;
  float* dblb = w; w += T * 32;
  float* biasb= w; w += (long)NLAY_ * NB_ * NS_ * 64 * 8;
  int*   albuf= (int*)w; w += T;
  float* qvec = w; w += 512;           // 2 x 256 (value, reward)
  float* obuf = w; w += BT * DM_;
  float* tvb  = w; w += BT * DM_;
  f16*   ah   = (f16*)w; w += T * DI_ / 2;        // packed A (f16), up to K=512
  f16*   wh16 = (f16*)w; w += (256 * 1024) / 2;   // packed W (f16), up to 256K elts

  auto blocks1d = [](long n) { return dim3((unsigned)((n + 255) / 256)); };
  // pack A (optional) + pack W + WMMA GEMM
  auto gemm = [&](const float* A, const float* Wf, const float* bias, const float* res,
                  float* C, long M, int N, int Kp, int Ko, int TW, int ACT, bool packA) {
    if (packA)
      k_cast16<<<blocks1d(M * Kp), 256, 0, stream>>>(A, ah, M * Kp);
    k_pack_w<<<blocks1d((long)N * Kp), 256, 0, stream>>>(Wf, wh16, N, Kp, Ko, TW);
    dim3 g((unsigned)(M / 64), (unsigned)((N + 127) / 128));
    k_gemm<<<g, 256, 0, stream>>>(ah, wh16, bias, res, C, (int)M, N, Kp, ACT);
  };

  // ---- relational trunk -> attention biases (all 4 layers at once) ----------
  k_trunk<<<dim3((unsigned)(NB_ * NS_ * 64)), 64, 0, stream>>>(
      pos, vel, P[P_WR1], P[P_BR1], P[P_WR2], P[P_BR2], P[P_AD_W], P[P_AD_B], biasb);

  // ---- embed: state encoder -> fusion ---------------------------------------
  k_state_enc<<<blocks1d(T * DM_), 256, 0, stream>>>(
      state, P[P_DEAD], P[P_SE_W1], P[P_SE_B1], albuf, tmpN);
  gemm(tmpN, P[P_SE_W2], P[P_SE_B2], nullptr, tmpD, T, DM_, DM_, DM_, 0, 0, true);
  k_build_cat<<<blocks1d(T * KCAT_), 256, 0, stream>>>(
      tmpD, team_ids, seq_idx, prev_action, P[P_TEAM_EMB], P[P_SHIP_EMB], P[P_RESET],
      P[P_AI_P], P[P_AI_T], P[P_AI_S], tmpA);
  gemm(tmpA, P[P_FUS_W], P[P_FUS_B], nullptr, tmpC, T, DM_, KCAT_, 304, 0, 0, true);
  k_norm<<<blocks1d(T * 32), 256, 0, stream>>>(tmpC, P[P_FUS_G], nullptr, nullptr, xm, T, 0, 1);

  // ---- 4 blocks: mamba + relational attention -------------------------------
  for (int l = 0; l < NLAY_; ++l) {
    k_norm<<<blocks1d(T * 32), 256, 0, stream>>>(xm, P[B_G1] + l * DM_, nullptr, nullptr, tmpN, T, 0, 0);
    gemm(tmpN, P[B_W_IN] + (long)l * DM_ * 1024, nullptr, nullptr, tmpA, T, 1024, DM_, DM_, 0, 0, true);
    k_conv<<<blocks1d(T * DI_), 256, 0, stream>>>(
        tmpA, P[B_CONV_W] + (long)l * DI_ * 4, P[B_CONV_B] + l * DI_, seq_idx, tmpB);
    gemm(tmpB, P[B_W_X] + (long)l * DI_ * 32, nullptr, nullptr, dblb, T, 32, DI_, DI_, 0, 0, true);
    k_dt<<<blocks1d(T * DI_), 256, 0, stream>>>(dblb, P[B_W_DT] + (long)l * 16 * DI_,
                                                P[B_B_DT] + l * DI_, tmpC);
    k_scan<<<dim3((NB_ * NN_ * DI_) / 256), 256, 0, stream>>>(
        tmpC, tmpB, dblb, tmpA, P[B_A_LOG] + (long)l * DI_ * DST_, P[B_D] + l * DI_,
        seq_idx, tmpD);
    gemm(tmpD, P[B_W_OUT] + (long)l * DI_ * DM_, nullptr, xm, xm, T, DM_, DI_, DI_, 0, 0, true);

    k_norm<<<blocks1d(T * 32), 256, 0, stream>>>(xm, P[B_G2] + l * DM_, nullptr, nullptr, tmpN, T, 0, 0);
    // one A-pack shared by Q/K/V projections
    gemm(tmpN, P[B_WQ] + (long)l * DM_ * DM_, P[B_BQ] + l * DM_, nullptr, qb, T, DM_, DM_, DM_, 0, 0, true);
    gemm(tmpN, P[B_WK] + (long)l * DM_ * DM_, P[B_BK] + l * DM_, nullptr, kb, T, DM_, DM_, DM_, 0, 0, false);
    gemm(tmpN, P[B_WV] + (long)l * DM_ * DM_, P[B_BV] + l * DM_, nullptr, vb, T, DM_, DM_, DM_, 0, 0, false);
    k_attn<<<blocks1d(NB_ * NS_ * 64), 256, 0, stream>>>(
        qb, kb, vb, biasb + (long)l * NB_ * NS_ * 64 * 8, albuf, tmpB);
    gemm(tmpB, P[B_WO] + (long)l * DM_ * DM_, P[B_BO] + l * DM_, xm, xm, T, DM_, DM_, DM_, 0, 0, true);
  }

  // ---- Z + action logits -----------------------------------------------------
  k_zout<<<blocks1d(T * DM_), 256, 0, stream>>>(xm, out);
  k_small<<<blocks1d(T * 12), 256, 0, stream>>>(xm, P[P_ACT_W], P[P_ACT_B], out + OFF_AL_, T, DM_, 12, 0, 1);

  // ---- value head (layernorm pool over ships) --------------------------------
  k_norm<<<blocks1d(T * 32), 256, 0, stream>>>(xm, P[P_LNV_W], P[P_LNV_B], nullptr, tmpN, T, 1, 0);
  k_poolq<<<1, 256, 0, stream>>>(P[P_TOK_VAL], P[PV_IN_W], P[PV_IN_B], qvec);
  gemm(tmpN, P[PV_IN_W] + 256 * 256, P[PV_IN_B] + 256, nullptr, kb, T, DM_, DM_, DM_, 1, 0, true);
  gemm(tmpN, P[PV_IN_W] + 512 * 256, P[PV_IN_B] + 512, nullptr, vb, T, DM_, DM_, DM_, 1, 0, false);
  k_pool_attn<<<blocks1d(BT * 4), 256, 0, stream>>>(qvec, kb, vb, albuf, obuf);
  gemm(obuf, P[PV_OUT_W], P[PV_OUT_B], nullptr, tvb, BT, DM_, DM_, DM_, 1, 0, true);
  gemm(tvb, P[P_VH_W1], P[P_VH_B1], nullptr, obuf, BT, DM_, DM_, DM_, 0, 1, true);
  k_small<<<blocks1d(BT), 256, 0, stream>>>(obuf, P[P_VH_W2], P[P_VH_B2], out + OFF_VAL_, BT, DM_, 1, 0, 0);

  // ---- dynamics path ---------------------------------------------------------
  k_dyn_cat<<<blocks1d(T * KCAT_), 256, 0, stream>>>(
      xm, out + OFF_AL_, P[P_ADN_P], P[P_ADN_T], P[P_ADN_S], tmpA);
  gemm(tmpA, P[P_DYN_W], P[P_DYN_B], nullptr, tmpC, T, DM_, KCAT_, 304, 0, 0, true);
  k_norm<<<blocks1d(T * 32), 256, 0, stream>>>(tmpC, P[P_DYN_G], nullptr, nullptr, qb, T, 0, 1); // x_dyn
  gemm(qb, P[P_FFN_W1], P[P_FFN_B1], nullptr, tmpB, T, DI_, DM_, DM_, 0, 1, true);
  gemm(tmpB, P[P_FFN_W2], P[P_FFN_B2], nullptr, tmpC, T, DM_, DI_, DI_, 0, 0, true);
  k_norm<<<blocks1d(T * 32), 256, 0, stream>>>(tmpC, P[P_FFN_G], nullptr, qb, vb, T, 0, 0);      // Zp

  // next_state
  gemm(vb, P[P_WH_W1], P[P_WH_B1], nullptr, tmpB, T, DM_, DM_, DM_, 0, 1, true);
  k_small<<<blocks1d(T * 5), 256, 0, stream>>>(tmpB, P[P_WH_W2], P[P_WH_B2], out + OFF_NS_, T, DM_, 5, 0, 1);

  // reward head (rms pool over ships on Zp)
  k_norm<<<blocks1d(T * 32), 256, 0, stream>>>(vb, P[P_RN_G], nullptr, nullptr, tmpN, T, 0, 0);
  k_poolq<<<1, 256, 0, stream>>>(P[P_TOK_REW], P[PR_IN_W], P[PR_IN_B], qvec + 256);
  gemm(tmpN, P[PR_IN_W] + 256 * 256, P[PR_IN_B] + 256, nullptr, kb, T, DM_, DM_, DM_, 1, 0, true);
  gemm(tmpN, P[PR_IN_W] + 512 * 256, P[PR_IN_B] + 512, nullptr, qb, T, DM_, DM_, DM_, 1, 0, false);
  k_pool_attn<<<blocks1d(BT * 4), 256, 0, stream>>>(qvec + 256, kb, qb, albuf, obuf);
  gemm(obuf, P[PR_OUT_W], P[PR_OUT_B], nullptr, tvb, BT, DM_, DM_, DM_, 1, 0, true);
  gemm(tvb, P[P_RH_W1], P[P_RH_B1], nullptr, obuf, BT, DM_, DM_, DM_, 0, 1, true);
  k_small<<<blocks1d(BT), 256, 0, stream>>>(obuf, P[P_RH_W2], P[P_RH_B2], out + OFF_REW_, BT, DM_, 1, 0, 0);
}
